// CustomQwen2Decoder_61048665145665
// MI455X (gfx1250) — matrix operand order repaired and checked by reading, not compiled, not run on hardware
//
#include <hip/hip_runtime.h>
#include <hip/hip_bf16.h>
#include <math.h>

// ---------------- model constants ----------------
#define DMODEL 896
#define NLAYER 24
#define HQ 14
#define HKV 2
#define DHEAD 64
#define IDIM 4864
#define EPSN 1e-6f
#define THETA 1000000.0f
#define BATCH 2
#define SEQ 2048
#define NTOK (BATCH * SEQ)

typedef __attribute__((ext_vector_type(16))) __bf16        v16bf;
typedef __attribute__((ext_vector_type(8)))  float         v8f;
typedef __attribute__((ext_vector_type(4)))  unsigned int  u32x4;
typedef __attribute__((ext_vector_type(2)))  unsigned int  u32x2;
typedef int v2i __attribute__((vector_size(8)));

union F16Q {
    v16bf          v;
    u32x4          q[2];
    unsigned short u[16];
};

#if __has_builtin(__builtin_amdgcn_global_load_async_to_lds_b64)
#define HAVE_ASYNC_LDS 1
#else
#define HAVE_ASYNC_LDS 0
#endif

#if HAVE_ASYNC_LDS
__device__ __forceinline__ void async_copy_b64(const void* g, void* l) {
    __builtin_amdgcn_global_load_async_to_lds_b64(
        (__attribute__((address_space(1))) v2i*)g,
        (__attribute__((address_space(3))) v2i*)l, 0, 0);
}
__device__ __forceinline__ void wait_async0() {
#if __has_builtin(__builtin_amdgcn_s_wait_asynccnt)
    __builtin_amdgcn_s_wait_asynccnt(0);
#else
    asm volatile("s_wait_asynccnt 0" ::: "memory");
#endif
}
#endif

__device__ __forceinline__ unsigned short f2bf(float f) {
    unsigned int u = __float_as_uint(f);
    u += 0x7fffu + ((u >> 16) & 1u);   // round-to-nearest-even
    return (unsigned short)(u >> 16);
}

// packed pair convert -> v_cvt_pk_bf16_f32
__device__ __forceinline__ unsigned int pack2bf(float a, float b) {
    __hip_bfloat162 h = __float22bfloat162_rn(make_float2(a, b));
    unsigned int r;
    __builtin_memcpy(&r, &h, sizeof(r));
    return r;
}

__device__ __forceinline__ v8f wmma_bf16(v16bf a, v16bf b, v8f c) {
    return __builtin_amdgcn_wmma_f32_16x16x32_bf16(false, a, false, b, (short)0, c, false, false);
}

// ---------------- RMSNorm (writes bf16 and/or f32) ----------------
__global__ __launch_bounds__(256) void rmsnorm_kernel(
    const float* __restrict__ x, const float* __restrict__ w,
    unsigned short* __restrict__ obf, float* __restrict__ of)
{
    __shared__ float red[8];
    int row = blockIdx.x;
    const float* xr = x + (size_t)row * DMODEL;
    float ss = 0.f;
    for (int i = threadIdx.x; i < DMODEL; i += blockDim.x) { float v = xr[i]; ss += v * v; }
    for (int off = 16; off; off >>= 1) ss += __shfl_down(ss, off, 32);
    int wid = threadIdx.x >> 5;
    if ((threadIdx.x & 31) == 0) red[wid] = ss;
    __syncthreads();
    if (threadIdx.x == 0) {
        float t = 0.f;
        for (int i = 0; i < 8; ++i) t += red[i];
        red[0] = t;
    }
    __syncthreads();
    float inv = rsqrtf(red[0] / (float)DMODEL + EPSN);
    for (int i = threadIdx.x; i < DMODEL; i += blockDim.x) {
        float v = xr[i] * inv * w[i];
        if (obf) obf[(size_t)row * DMODEL + i] = f2bf(v);
        if (of)  of[(size_t)row * DMODEL + i] = v;
    }
}

// ---------------- bf16 WMMA GEMM:  C(MxN) = A(MxK,bf16) * B(KxN,f32 weights) ----------------
#define BM 256
#define BN 64
#define BK 32
#define ASTR 40            // As row stride in shorts (80B, multiple of 16B)
#define EPI_ADD  1
#define EPI_SILU 2

template <int FLAGS, bool HASBIAS>
__global__ __launch_bounds__(256) void gemm_bf16_kernel(
    const unsigned short* __restrict__ A, int lda,
    const float* __restrict__ Bw, int ldb,
    const float* __restrict__ bias,
    float* __restrict__ C, int ldc,
    int K)
{
    __shared__ unsigned short As[BM][ASTR];          // row-major A tile
    __shared__ unsigned short Bfrag[4][32][16];      // B staged in WMMA fragment order

    int tid  = threadIdx.x;
    int n0   = blockIdx.x * BN;
    int m0   = blockIdx.y * BM;
    int wv   = tid >> 5;
    int lane = tid & 31;
    int lm   = lane & 15;
    int lh   = lane >> 4;

    v8f acc[2][4];
#pragma unroll
    for (int s = 0; s < 2; ++s)
#pragma unroll
        for (int t = 0; t < 4; ++t)
#pragma unroll
            for (int e = 0; e < 8; ++e) acc[s][t][e] = 0.f;

    // ---- hoisted staging pointers (walked, not re-derived) ----
    int ar  = tid >> 3;            // A: row group 0..31
    int ac4 = (tid & 7) * 4;       // A: col 0..28
    const unsigned short* aptr = A + (size_t)(m0 + ar) * lda + ac4;
    const size_t astep = (size_t)32 * lda;     // 32 rows per pass

    int bn  = tid & 63;            // B: column (coalesced across lanes)
    int bkq = (tid >> 6) * 8;      // B: k-group base {0,8,16,24}
    int bt  = bn >> 4;             // B: 16-wide N tile
    int bl0 = bn & 15;             // lane low bits
    const float* bptr = Bw + (size_t)bkq * ldb + n0 + bn;
    unsigned short* bdst = &Bfrag[bt][2 * (bkq & 8) + bl0][(bkq & 16) >> 1];

    for (int k0 = 0; k0 < K; k0 += BK) {
        // stage A tile: 256x32 bf16
#if HAVE_ASYNC_LDS
#pragma unroll
        for (int p = 0; p < 8; ++p)
            async_copy_b64(aptr + p * astep, &As[p * 32 + ar][ac4]);
#else
#pragma unroll
        for (int p = 0; p < 8; ++p) {
            uint2 d = *(const uint2*)(aptr + p * astep);
            *(uint2*)&As[p * 32 + ar][ac4] = d;
        }
#endif
        // stage B tile directly in fragment order:
        // element (k,n) -> Bfrag[n/16][2*(k&8)+(n&15)][(k&7)+((k&16)>>1)]
#pragma unroll
        for (int p = 0; p < 2; ++p) {
            const float* src = bptr + (size_t)(p * 4) * ldb;
            float v0 = src[0];
            float v1 = src[(size_t)ldb];
            float v2 = src[(size_t)ldb * 2];
            float v3 = src[(size_t)ldb * 3];
            u32x2 d;
            d.x = pack2bf(v0, v1);
            d.y = pack2bf(v2, v3);
            *(u32x2*)(bdst + p * 4) = d;
        }
        aptr += BK;
        bptr += (size_t)BK * ldb;
#if HAVE_ASYNC_LDS
        wait_async0();
#endif
        __syncthreads();

        // A fragments for this wave's two 16-row sub-tiles (b128 loads)
        F16Q af0, af1;
        int koff = lh * 8;
        int r0 = wv * 32 + lm;
        af0.q[0] = *(const u32x4*)&As[r0][koff];
        af0.q[1] = *(const u32x4*)&As[r0][16 + koff];
        af1.q[0] = *(const u32x4*)&As[r0 + 16][koff];
        af1.q[1] = *(const u32x4*)&As[r0 + 16][16 + koff];
#pragma unroll
        for (int t = 0; t < 4; ++t) {
            F16Q bf;
            bf.q[0] = *(const u32x4*)&Bfrag[t][lane][0];
            bf.q[1] = *(const u32x4*)&Bfrag[t][lane][8];
            acc[0][t] = wmma_bf16(af0.v, bf.v, acc[0][t]);
            acc[1][t] = wmma_bf16(af1.v, bf.v, acc[1][t]);
        }
        __syncthreads();
    }

    // epilogue (branch-free, pointer-walked)
    int mb = lh * 8;
    float bcol[4];
#pragma unroll
    for (int t = 0; t < 4; ++t) bcol[t] = HASBIAS ? bias[n0 + t * 16 + lm] : 0.f;
    float* cbase = C + (size_t)(m0 + wv * 32 + mb) * ldc + n0 + lm;
#pragma unroll
    for (int s = 0; s < 2; ++s) {
#pragma unroll
        for (int t = 0; t < 4; ++t) {
            float* cp = cbase + (size_t)(s * 16) * ldc + t * 16;
#pragma unroll
            for (int v = 0; v < 8; ++v) {
                float val = acc[s][t][v] + bcol[t];
                if (FLAGS & EPI_SILU) val = val / (1.f + __expf(-val));
                if (FLAGS & EPI_ADD)  val += cp[0];
                cp[0] = val;
                cp += ldc;
            }
        }
    }
}

// ---------------- RoPE (in place, q or k buffer) ----------------
__global__ __launch_bounds__(256) void rope_kernel(float* __restrict__ qk, int nheads, int total)
{
    int idx = blockIdx.x * blockDim.x + threadIdx.x;
    if (idx >= total) return;
    int i = idx % (DHEAD / 2);
    int h = (idx / (DHEAD / 2)) % nheads;
    int t = idx / ((DHEAD / 2) * nheads);
    int s = t % SEQ;
    float invf = __expf(-(2.0f * (float)i / (float)DHEAD) * logf(THETA));
    float ang = (float)s * invf;
    float c = cosf(ang), sn = sinf(ang);
    size_t base = (size_t)t * ((size_t)nheads * DHEAD) + (size_t)h * DHEAD;
    float x1 = qk[base + i];
    float x2 = qk[base + i + DHEAD / 2];
    qk[base + i]             = x1 * c - x2 * sn;
    qk[base + i + DHEAD / 2] = x2 * c + x1 * sn;
}

// ---------------- flash attention with WMMA ----------------
#define MNEG (-3.0e38f)
#define KSTR 72            // Kc row stride in shorts (144B, multiple of 16B)
#define VSTR 40            // Vct row stride in shorts (80B, multiple of 16B)
#define PSTR 40            // Pst row stride in shorts

__global__ __launch_bounds__(128) void attn_kernel(
    const float* __restrict__ qb, const float* __restrict__ kb, const float* __restrict__ vb,
    const int* __restrict__ tt, const float* __restrict__ am,
    unsigned short* __restrict__ obf)
{
    __shared__ unsigned short Kc[32][KSTR];        // [key][dh]
    __shared__ unsigned short Vct[DHEAD][VSTR];    // [dh][key]  (transposed V)
    __shared__ unsigned short Pst[4][16][PSTR];    // per-wave P restage
    __shared__ int   ttq[64];
    __shared__ int   ttk[32];
    __shared__ float pkm[32];

    int qblk = blockIdx.x, h = blockIdx.y, b = blockIdx.z;
    int kvh  = h / (HQ / HKV);
    int tid  = threadIdx.x;
    int wv   = tid >> 5;
    int lane = tid & 31;
    int lm   = lane & 15;
    int lh   = lane >> 4;
    int q0   = qblk * 64;       // block's base query (within sequence)
    int qt   = q0 + wv * 16;    // this wave's query tile base

    // Q fragments (A layout) for the two 32-wide head-dim halves, float4 loads
    F16Q aq0, aq1;
    {
        const float* qr = qb + ((size_t)(b * SEQ + qt + lm)) * (HQ * DHEAD) + (size_t)h * DHEAD;
        int koff = lh * 8;
        float4 d0 = *(const float4*)(qr + koff);
        float4 d1 = *(const float4*)(qr + koff + 4);
        float4 d2 = *(const float4*)(qr + 16 + koff);
        float4 d3 = *(const float4*)(qr + 16 + koff + 4);
        float4 d4 = *(const float4*)(qr + 32 + koff);
        float4 d5 = *(const float4*)(qr + 32 + koff + 4);
        float4 d6 = *(const float4*)(qr + 48 + koff);
        float4 d7 = *(const float4*)(qr + 48 + koff + 4);
        aq0.q[0] = (u32x4){pack2bf(d0.x, d0.y), pack2bf(d0.z, d0.w), pack2bf(d1.x, d1.y), pack2bf(d1.z, d1.w)};
        aq0.q[1] = (u32x4){pack2bf(d2.x, d2.y), pack2bf(d2.z, d2.w), pack2bf(d3.x, d3.y), pack2bf(d3.z, d3.w)};
        aq1.q[0] = (u32x4){pack2bf(d4.x, d4.y), pack2bf(d4.z, d4.w), pack2bf(d5.x, d5.y), pack2bf(d5.z, d5.w)};
        aq1.q[1] = (u32x4){pack2bf(d6.x, d6.y), pack2bf(d6.z, d6.w), pack2bf(d7.x, d7.y), pack2bf(d7.z, d7.w)};
    }
    if (tid < 64) ttq[tid] = tt[b * SEQ + q0 + tid];
    __syncthreads();

    float mi[8], li[8];
    v8f oacc[4];
#pragma unroll
    for (int v = 0; v < 8; ++v) { mi[v] = MNEG; li[v] = 0.f; }
#pragma unroll
    for (int t = 0; t < 4; ++t)
#pragma unroll
        for (int e = 0; e < 8; ++e) oacc[t][e] = 0.f;

    const float scale = 0.125f;  // 1/sqrt(64)

    // hoisted chunk pointers (walked by 32 keys per iteration)
    int kr  = tid >> 4;          // K: 0..7
    int kc4 = (tid & 15) * 4;    // K: dh col 0..60
    int vdh = tid & 63;          // V: dh (coalesced)
    int vkq = (tid >> 6) * 16;   // V: key group {0,16}
    const float* kptr = kb + ((size_t)(b * SEQ) + kr) * (HKV * DHEAD) + (size_t)kvh * DHEAD + kc4;
    const float* vptr = vb + ((size_t)(b * SEQ) + vkq) * (HKV * DHEAD) + (size_t)kvh * DHEAD + vdh;
    const int*   ttp  = tt + b * SEQ;
    const float* amp  = am + b * SEQ;
    const size_t chunkstep = (size_t)32 * (HKV * DHEAD);

    for (int kc = 0; kc < SEQ; kc += 32) {
        // stage K chunk row-major (b64 stores)
#pragma unroll
        for (int p = 0; p < 4; ++p) {
            float4 dk = *(const float4*)(kptr + (size_t)(p * 8) * (HKV * DHEAD));
            u32x2 d;
            d.x = pack2bf(dk.x, dk.y);
            d.y = pack2bf(dk.z, dk.w);
            *(u32x2*)&Kc[p * 8 + kr][kc4] = d;
        }
        // stage V chunk transposed: Vct[dh][key] (b64 stores)
#pragma unroll
        for (int p = 0; p < 4; ++p) {
            const float* src = vptr + (size_t)(p * 4) * (HKV * DHEAD);
            float v0 = src[0];
            float v1 = src[HKV * DHEAD];
            float v2 = src[HKV * DHEAD * 2];
            float v3 = src[HKV * DHEAD * 3];
            u32x2 d;
            d.x = pack2bf(v0, v1);
            d.y = pack2bf(v2, v3);
            *(u32x2*)&Vct[vdh][vkq + p * 4] = d;
        }
        if (tid < 32) {
            ttk[tid] = ttp[kc + tid];
            pkm[tid] = amp[kc + tid];
        }
        kptr += chunkstep;
        vptr += chunkstep;
        __syncthreads();

        // scores: S(16q x 32k) = Q(16x64) @ K^T(64x32); B-fragments contiguous along dh
        v8f s0, s1;
#pragma unroll
        for (int e = 0; e < 8; ++e) { s0[e] = 0.f; s1[e] = 0.f; }
        int kbq = lh * 8;
        F16Q b00, b01, b10, b11;
        b00.q[0] = *(const u32x4*)&Kc[lm][kbq];
        b00.q[1] = *(const u32x4*)&Kc[lm][kbq + 16];
        b01.q[0] = *(const u32x4*)&Kc[lm][32 + kbq];
        b01.q[1] = *(const u32x4*)&Kc[lm][32 + kbq + 16];
        b10.q[0] = *(const u32x4*)&Kc[16 + lm][kbq];
        b10.q[1] = *(const u32x4*)&Kc[16 + lm][kbq + 16];
        b11.q[0] = *(const u32x4*)&Kc[16 + lm][32 + kbq];
        b11.q[1] = *(const u32x4*)&Kc[16 + lm][32 + kbq + 16];
        s0 = wmma_bf16(aq0.v, b00.v, s0);
        s0 = wmma_bf16(aq1.v, b01.v, s0);
        s1 = wmma_bf16(aq0.v, b10.v, s1);
        s1 = wmma_bf16(aq1.v, b11.v, s1);

        // bias + online softmax (rows live in 16-lane halves)
        int mb = lh * 8;
        float p0[8], p1[8], alpha[8];
#pragma unroll
        for (int v = 0; v < 8; ++v) {
            int m  = mb + v;
            int qg = q0 + wv * 16 + m;
            int tq = ttq[wv * 16 + m];
            int kg0 = kc + lm, kg1 = kc + 16 + lm;
            int tk0 = ttk[lm], tk1 = ttk[16 + lm];
            bool a0 = (((tq == 0) && (tk0 == 0)) ||
                       ((tq == 1) && ((tk0 == 0) || ((tk0 == 1) && (kg0 <= qg))))) && (pkm[lm] > 0.5f);
            bool a1 = (((tq == 0) && (tk1 == 0)) ||
                       ((tq == 1) && ((tk1 == 0) || ((tk1 == 1) && (kg1 <= qg))))) && (pkm[16 + lm] > 0.5f);
            float sv0 = a0 ? s0[v] * scale : MNEG;
            float sv1 = a1 ? s1[v] * scale : MNEG;
            float mx = fmaxf(sv0, sv1);
#pragma unroll
            for (int off = 8; off; off >>= 1) mx = fmaxf(mx, __shfl_xor(mx, off, 32));
            float mnew = fmaxf(mi[v], mx);
            float al = __expf(mi[v] - mnew);
            float e0 = __expf(sv0 - mnew);
            float e1 = __expf(sv1 - mnew);
            float ls = e0 + e1;
#pragma unroll
            for (int off = 8; off; off >>= 1) ls += __shfl_xor(ls, off, 32);
            li[v] = li[v] * al + ls;
            mi[v] = mnew;
            alpha[v] = al;
            p0[v] = e0;
            p1[v] = e1;
        }
#pragma unroll
        for (int t = 0; t < 4; ++t)
#pragma unroll
            for (int v = 0; v < 8; ++v) oacc[t][v] *= alpha[v];

        // restage P (C-layout f32) into A-layout bf16 via LDS
#pragma unroll
        for (int v = 0; v < 8; ++v) {
            Pst[wv][mb + v][lm]      = f2bf(p0[v]);
            Pst[wv][mb + v][16 + lm] = f2bf(p1[v]);
        }
        __syncthreads();

        F16Q ap;
        int koff = lh * 8;
        ap.q[0] = *(const u32x4*)&Pst[wv][lm][koff];
        ap.q[1] = *(const u32x4*)&Pst[wv][lm][16 + koff];
#pragma unroll
        for (int t = 0; t < 4; ++t) {
            F16Q bvv;
            bvv.q[0] = *(const u32x4*)&Vct[t * 16 + lm][kbq];
            bvv.q[1] = *(const u32x4*)&Vct[t * 16 + lm][kbq + 16];
            oacc[t] = wmma_bf16(ap.v, bvv.v, oacc[t]);
        }
        __syncthreads();
    }

    // finalize: O /= l, emit bf16 in (token, h*64+d) layout feeding the wo GEMM
#pragma unroll
    for (int t = 0; t < 4; ++t)
#pragma unroll
        for (int v = 0; v < 8; ++v) {
            int m = lh * 8 + v;
            float val = oacc[t][v] / li[v];
            size_t oi = ((size_t)(b * SEQ + q0 + wv * 16 + m)) * (HQ * DHEAD)
                      + (size_t)h * DHEAD + t * 16 + lm;
            obf[oi] = f2bf(val);
        }
}

// ---------------- elementwise: m = bf16(silu_g * u), vectorized x4 ----------------
__global__ __launch_bounds__(256) void mulbf_kernel(
    const float4* __restrict__ g, const float4* __restrict__ u,
    u32x2* __restrict__ o, int n4)
{
    int i = blockIdx.x * blockDim.x + threadIdx.x;
    if (i < n4) {
        float4 a = g[i];
        float4 b = u[i];
        u32x2 d;
        d.x = pack2bf(a.x * b.x, a.y * b.y);
        d.y = pack2bf(a.z * b.z, a.w * b.w);
        o[i] = d;
    }
}

// ---------------- host-side orchestration ----------------
extern "C" void kernel_launch(void* const* d_in, const int* in_sizes, int n_in,
                              void* d_out, int out_size, void* d_ws, size_t ws_size,
                              hipStream_t stream)
{
    const float* embeds = (const float*)d_in[0];
    const int*   tt     = (const int*)d_in[1];
    const float* am     = (const float*)d_in[2];
    const float* wq = (const float*)d_in[3];
    const float* bq = (const float*)d_in[4];
    const float* wk = (const float*)d_in[5];
    const float* bk = (const float*)d_in[6];
    const float* wv = (const float*)d_in[7];
    const float* bv = (const float*)d_in[8];
    const float* wo = (const float*)d_in[9];
    const float* wg = (const float*)d_in[10];
    const float* wu = (const float*)d_in[11];
    const float* wd = (const float*)d_in[12];
    const float* ln1 = (const float*)d_in[13];
    const float* ln2 = (const float*)d_in[14];
    const float* lnf = (const float*)d_in[15];

    char* ws = (char*)d_ws;
    size_t off = 0;
    auto alloc = [&](size_t bytes) -> void* {
        void* p = ws + off;
        off += (bytes + 255) & ~(size_t)255;
        return p;
    };
    float*          x    = (float*)alloc((size_t)NTOK * DMODEL * 4);
    unsigned short* hbf  = (unsigned short*)alloc((size_t)NTOK * DMODEL * 2);
    float*          qbuf = (float*)alloc((size_t)NTOK * HQ * DHEAD * 4);
    float*          kbuf = (float*)alloc((size_t)NTOK * HKV * DHEAD * 4);
    float*          vbuf = (float*)alloc((size_t)NTOK * HKV * DHEAD * 4);
    unsigned short* obf  = (unsigned short*)alloc((size_t)NTOK * HQ * DHEAD * 2);
    float*          gbuf = (float*)alloc((size_t)NTOK * IDIM * 4);
    float*          ubuf = (float*)alloc((size_t)NTOK * IDIM * 4);
    unsigned short* mbf  = (unsigned short*)alloc((size_t)NTOK * IDIM * 2);
    (void)ws_size; (void)in_sizes; (void)n_in; (void)out_size;

    (void)hipMemcpyAsync(x, embeds, (size_t)NTOK * DMODEL * 4, hipMemcpyDeviceToDevice, stream);

    dim3 gemmGridQ(DMODEL / BN, NTOK / BM);        // N=896
    dim3 gemmGridKV(128 / BN, NTOK / BM);          // N=128
    dim3 gemmGridI(IDIM / BN, NTOK / BM);          // N=4864

    for (int l = 0; l < NLAYER; ++l) {
        // h = rmsnorm(x, ln1)  (bf16)
        rmsnorm_kernel<<<NTOK, 256, 0, stream>>>(x, ln1 + (size_t)l * DMODEL, hbf, nullptr);

        // q/k/v projections (+bias)
        gemm_bf16_kernel<0, true><<<gemmGridQ, 256, 0, stream>>>(
            hbf, DMODEL, wq + (size_t)l * DMODEL * (HQ * DHEAD), HQ * DHEAD,
            bq + (size_t)l * (HQ * DHEAD), qbuf, HQ * DHEAD, DMODEL);
        gemm_bf16_kernel<0, true><<<gemmGridKV, 256, 0, stream>>>(
            hbf, DMODEL, wk + (size_t)l * DMODEL * (HKV * DHEAD), HKV * DHEAD,
            bk + (size_t)l * (HKV * DHEAD), kbuf, HKV * DHEAD, DMODEL);
        gemm_bf16_kernel<0, true><<<gemmGridKV, 256, 0, stream>>>(
            hbf, DMODEL, wv + (size_t)l * DMODEL * (HKV * DHEAD), HKV * DHEAD,
            bv + (size_t)l * (HKV * DHEAD), vbuf, HKV * DHEAD, DMODEL);

        // RoPE on q and k
        {
            int tq = NTOK * HQ * (DHEAD / 2);
            int tk = NTOK * HKV * (DHEAD / 2);
            rope_kernel<<<(tq + 255) / 256, 256, 0, stream>>>(qbuf, HQ, tq);
            rope_kernel<<<(tk + 255) / 256, 256, 0, stream>>>(kbuf, HKV, tk);
        }

        // attention (flash, GQA, token-type bias)
        attn_kernel<<<dim3(SEQ / 64, HQ, BATCH), 128, 0, stream>>>(
            qbuf, kbuf, vbuf, tt, am, obf);

        // x += o @ wo
        gemm_bf16_kernel<EPI_ADD, false><<<gemmGridQ, 256, 0, stream>>>(
            obf, HQ * DHEAD, wo + (size_t)l * (HQ * DHEAD) * DMODEL, DMODEL,
            nullptr, x, DMODEL, HQ * DHEAD);

        // h2 = rmsnorm(x, ln2)  (bf16)
        rmsnorm_kernel<<<NTOK, 256, 0, stream>>>(x, ln2 + (size_t)l * DMODEL, hbf, nullptr);

        // gate (with SiLU) and up
        gemm_bf16_kernel<EPI_SILU, false><<<gemmGridI, 256, 0, stream>>>(
            hbf, DMODEL, wg + (size_t)l * DMODEL * IDIM, IDIM,
            nullptr, gbuf, IDIM, DMODEL);
        gemm_bf16_kernel<0, false><<<gemmGridI, 256, 0, stream>>>(
            hbf, DMODEL, wu + (size_t)l * DMODEL * IDIM, IDIM,
            nullptr, ubuf, IDIM, DMODEL);

        // m = bf16(silu(g) * u), 4 elements per thread
        {
            int n4 = (NTOK * IDIM) / 4;
            mulbf_kernel<<<(n4 + 255) / 256, 256, 0, stream>>>(
                (const float4*)gbuf, (const float4*)ubuf, (u32x2*)mbf, n4);
        }

        // x += m @ wd
        gemm_bf16_kernel<EPI_ADD, false><<<gemmGridQ, 256, 0, stream>>>(
            mbf, IDIM, wd + (size_t)l * IDIM * DMODEL, DMODEL,
            nullptr, x, DMODEL, IDIM);
    }

    // final norm -> f32 output
    rmsnorm_kernel<<<NTOK, 256, 0, stream>>>(x, lnf, nullptr, (float*)d_out);
}
